// MyResnetGate_20529943675342
// MI455X (gfx1250) — matrix operand order repaired and checked by reading, not compiled, not run on hardware
//
#include <hip/hip_runtime.h>

typedef __attribute__((ext_vector_type(16))) _Float16 v16h;
typedef __attribute__((ext_vector_type(8)))  float    v8f;

union Frag16 { v16h h; unsigned u[8]; uint4 q[2]; };

// single v_cvt_pk_rtz_f16_f32
static __device__ __forceinline__ unsigned pk2(float a, float b) {
  auto v = __builtin_amdgcn_cvt_pkrtz(a, b);   // __fp16 ext_vector_type(2)
  return __builtin_bit_cast(unsigned, v);
}

// ---------------------------------------------------------------------------
// Kernel 1: batched 1x1 conv as GEMM.  out[b,n,p] = sum_c x[b,c,p]*W[n,c] + bias[n]
// View: M = p (spatial), N = n (out ch), K = c (in ch).
// Block = 128 threads (4 waves). Tile BM=16, BN=256 (wave handles 16x64).
// Addressing: uniform base (W + k0 / xb + k0*P) + per-thread 32-bit offsets.
// ---------------------------------------------------------------------------
__global__ __launch_bounds__(128)
void conv1x1_wmma(const float* __restrict__ x, const float* __restrict__ W,
                  const float* __restrict__ bias, float* __restrict__ out,
                  int C, int P, int Nch) {
  __shared__ __align__(16) _Float16 Ash[16][32];   // [m][k]  (k-pairs adjacent)
  __shared__ __align__(16) _Float16 Bsh[256][32];  // [n][k]

  const int t    = threadIdx.x;
  const int wave = t >> 5;
  const int lane = t & 31;
  const int m16  = lane & 15;
  const int hi   = lane >> 4;

  const int b      = blockIdx.z;
  const int mbase  = blockIdx.x * 16;
  const int nblock = blockIdx.y * 256;

  const float* xb = x + (size_t)b * C * P;

  // A staging role (fixed per thread): row m, k-quad kb..kb+3 (strided loads)
  const int am  = t & 15;
  const int akb = (t >> 4) * 4;
  const int amg = mbase + am;
  const int amc = min(amg, P - 1);
  const bool aok = (amg < P);
  const int aoff0 = akb * P + amc;       // per-thread, loop-invariant
  const int aoff1 = aoff0 + P;
  const int aoff2 = aoff1 + P;
  const int aoff3 = aoff2 + P;

  // B staging offsets (per-thread, loop-invariant 32-bit element offsets)
  int boff[16];
  #pragma unroll
  for (int j = 0; j < 16; ++j) {
    int gi = j * 128 + t;                // 0..2047 granules of 4 floats
    int nl = gi >> 3;                    // 8 granules per 32-k row
    int kq = (gi & 7) * 4;
    boff[j] = (nblock + nl) * C + kq;
  }

  v8f acc[4] = {};

  const int ksteps = C >> 5;
  float av0, av1, av2, av3;
  {
    av0 = xb[aoff0]; av1 = xb[aoff1]; av2 = xb[aoff2]; av3 = xb[aoff3];
  }

  for (int ks = 0; ks < ksteps; ++ks) {
    const int k0 = ks << 5;
    const float* Wk = W + k0;            // uniform (SGPR) base

    // --- commit prefetched A quad (branch-free OOB select) ---
    {
      uint2 pkd;
      pkd.x = pk2(aok ? av0 : 0.0f, aok ? av1 : 0.0f);
      pkd.y = pk2(aok ? av2 : 0.0f, aok ? av3 : 0.0f);
      *(uint2*)&Ash[am][akb] = pkd;
    }
    // --- stage B tile in 2 groups of 8 batched b128 loads ---
    #pragma unroll
    for (int g = 0; g < 2; ++g) {
      float4 w4[8];
      #pragma unroll
      for (int jj = 0; jj < 8; ++jj)
        w4[jj] = *(const float4*)(Wk + boff[g * 8 + jj]);
      #pragma unroll
      for (int jj = 0; jj < 8; ++jj) {
        int gi = (g * 8 + jj) * 128 + t;
        int nl = gi >> 3;
        int kq = (gi & 7) * 4;
        uint2 pkd;
        pkd.x = pk2(w4[jj].x, w4[jj].y);
        pkd.y = pk2(w4[jj].z, w4[jj].w);
        *(uint2*)&Bsh[nl][kq] = pkd;
      }
    }
    __syncthreads();

    // prefetch next A quad while WMMAs run (uniform base advances by 32*P)
    if (ks + 1 < ksteps) {
      const float* xk = xb + (size_t)(k0 + 32) * P;
      av0 = xk[aoff0]; av1 = xk[aoff1]; av2 = xk[aoff2]; av3 = xk[aoff3];
    }

    // A fragment: lane m=m16; halves [hi*8 .. +7] and [16+hi*8 .. +7]
    Frag16 a;
    a.q[0] = *(const uint4*)&Ash[m16][hi * 8];
    a.q[1] = *(const uint4*)&Ash[m16][16 + hi * 8];

    #pragma unroll
    for (int tn = 0; tn < 4; ++tn) {
      // B fragment: lane n = tile_base + m16; halves [hi*8 .. +7] and [16+hi*8 .. +7]
      Frag16 bf;
      int nl = wave * 64 + tn * 16 + m16;
      bf.q[0] = *(const uint4*)&Bsh[nl][hi * 8];
      bf.q[1] = *(const uint4*)&Bsh[nl][16 + hi * 8];
      acc[tn] = __builtin_amdgcn_wmma_f32_16x16x32_f16(
          false, a.h, false, bf.h, (short)0, acc[tn], false, false);
    }
    __syncthreads();
  }

  // Epilogue: C/D layout: lane n = m16, VGPR r -> m = r + 8*hi
  #pragma unroll
  for (int tn = 0; tn < 4; ++tn) {
    int n = nblock + wave * 64 + tn * 16 + m16;
    float bv = bias[n];
    #pragma unroll
    for (int r = 0; r < 8; ++r) {
      int m = mbase + r + hi * 8;
      if (m < P) out[((size_t)b * Nch + n) * P + m] = acc[tn][r] + bv;
    }
  }
}

// ---------------------------------------------------------------------------
// Kernel 2: align-corners bilinear resize of f4 (14x14,512ch) and f5 (7x7,1024ch)
// to 28x28 and concat -> app (8,1536,28,28) fp32
// ---------------------------------------------------------------------------
__global__ void resize_concat(const float* __restrict__ f4,
                              const float* __restrict__ f5,
                              float* __restrict__ app, int total) {
  int idx = blockIdx.x * blockDim.x + threadIdx.x;
  if (idx >= total) return;
  int ow = idx % 28;
  int tt = idx / 28;
  int oh = tt % 28;  tt /= 28;
  int c  = tt % 1536;
  int b  = tt / 1536;

  const float* src;
  int H;
  if (c < 512) { src = f4 + ((size_t)b * 512 + c) * 196;         H = 14; }
  else         { src = f5 + ((size_t)b * 1024 + (c - 512)) * 49; H = 7;  }

  float scale = (float)(H - 1) / 27.0f;
  float y = oh * scale, x = ow * scale;
  int y0 = (int)floorf(y); int y1 = min(y0 + 1, H - 1);
  int x0 = (int)floorf(x); int x1 = min(x0 + 1, H - 1);
  float ly = y - (float)y0, lx = x - (float)x0;
  float v00 = src[y0 * H + x0], v01 = src[y0 * H + x1];
  float v10 = src[y1 * H + x0], v11 = src[y1 * H + x1];
  app[idx] = v00 * (1.f - ly) * (1.f - lx) + v01 * (1.f - ly) * lx +
             v10 * ly * (1.f - lx)         + v11 * ly * lx;
}

// ---------------------------------------------------------------------------
// Kernel 3: per-(b,g) gate + ROI-align(5x5) + LayerNorm(D=38400) + ReLU -> f16
// One block per row (64 rows).  Dynamic LDS: 38400 floats (150 KB < 320 KB/WG).
// ---------------------------------------------------------------------------
__global__ __launch_bounds__(256)
void roi_ln1(const float* __restrict__ app, const float* __restrict__ boxes,
             const float* __restrict__ skeleton, const int* __restrict__ group,
             const long long* __restrict__ label_s,
             const float* __restrict__ c1w, const float* __restrict__ c1b,
             const float* __restrict__ n1w, const float* __restrict__ n1b,
             _Float16* __restrict__ sk) {
  extern __shared__ float vals[];         // 38400 floats
  __shared__ float cw[25][4];
  __shared__ int   cix[25][4];
  __shared__ float red[256];
  __shared__ float s_mu, s_rstd;

  const int r = blockIdx.x;               // 0..63  (= b*8 + g)
  const int b = r >> 3;
  const int t = threadIdx.x;
  const int D = 38400;

  if (t < 25) {
    int i1 = group[r * 2 + 0];
    int i2 = group[r * 2 + 1];
    const float* bb = boxes + (size_t)b * 16 * 4;
    float ux1 = fminf(bb[i1 * 4 + 0], bb[i2 * 4 + 0]);
    float uy1 = fminf(bb[i1 * 4 + 1], bb[i2 * 4 + 1]);
    float ux2 = fmaxf(bb[i1 * 4 + 2], bb[i2 * 4 + 2]);
    float uy2 = fmaxf(bb[i1 * 4 + 3], bb[i2 * 4 + 3]);
    float rw = fmaxf(ux2 - ux1, 1.0f);
    float rh = fmaxf(uy2 - uy1, 1.0f);
    bool  ug = (label_s[r] != -1LL);
    float w1 = c1w[0], bg = c1b[0];
    const float* skel = skeleton + (size_t)r * 784;

    int ph = t / 5, pw = t % 5;
    float ys = uy1 + ((float)ph + 0.5f) * rh * 0.2f;
    float xs = ux1 + ((float)pw + 0.5f) * rw * 0.2f;
    float valid = (ys >= -1.f && ys <= 28.f && xs >= -1.f && xs <= 28.f) ? 1.f : 0.f;
    float y = fminf(fmaxf(ys, 0.f), 27.f);
    float x = fminf(fmaxf(xs, 0.f), 27.f);
    int y0 = (int)floorf(y), x0 = (int)floorf(x);
    int y1i = min(y0 + 1, 27), x1i = min(x0 + 1, 27);
    float ly = y - (float)y0, lx = x - (float)x0;
    float wts[4] = { (1.f - ly) * (1.f - lx), (1.f - ly) * lx,
                     ly * (1.f - lx),          ly * lx };
    int yy[4] = { y0, y0, y1i, y1i };
    int xx[4] = { x0, x1i, x0, x1i };
    #pragma unroll
    for (int j = 0; j < 4; ++j) {
      int off = yy[j] * 28 + xx[j];
      float gf = 1.0f;
      if (ug) gf = 1.0f / (1.0f + expf(-(skel[off] * w1 + bg)));
      cw[t][j]  = wts[j] * gf * valid;
      cix[t][j] = off;
    }
  }
  __syncthreads();

  const float* ab = app + (size_t)b * 1536 * 784;
  float sum = 0.f, ssq = 0.f;
  for (int i = t; i < D; i += 256) {
    int c = i / 25, p = i % 25;
    const float* pl = ab + (size_t)c * 784;
    float v = cw[p][0] * pl[cix[p][0]] + cw[p][1] * pl[cix[p][1]] +
              cw[p][2] * pl[cix[p][2]] + cw[p][3] * pl[cix[p][3]];
    vals[i] = v;
    sum += v;
    ssq += v * v;
  }

  red[t] = sum; __syncthreads();
  for (int s = 128; s > 0; s >>= 1) { if (t < s) red[t] += red[t + s]; __syncthreads(); }
  if (t == 0) s_mu = red[0] / (float)D;
  __syncthreads();
  red[t] = ssq; __syncthreads();
  for (int s = 128; s > 0; s >>= 1) { if (t < s) red[t] += red[t + s]; __syncthreads(); }
  if (t == 0) {
    float mu  = s_mu;
    float var = red[0] / (float)D - mu * mu;
    s_rstd = rsqrtf(var + 1e-5f);
  }
  __syncthreads();

  float mu = s_mu, rstd = s_rstd;
  _Float16* skr = sk + (size_t)r * D;
  for (int i = t; i < D; i += 256) {
    float v = (vals[i] - mu) * rstd * n1w[i] + n1b[i];
    skr[i] = (_Float16)fmaxf(v, 0.f);
  }
}

// ---------------------------------------------------------------------------
// Kernel 4: fc1 GEMM, M=64 N=512 K=38400, split-K with f32 atomic reduction.
// Block = 4 waves; wave w owns M-tile w (all of M covered) x 64 N.
// B staging software-pipelined; all addressing uniform-base + 32-bit offsets.
// ---------------------------------------------------------------------------
__global__ __launch_bounds__(128)
void fc1_wmma(const _Float16* __restrict__ sk, const float* __restrict__ W,
              float* __restrict__ h1) {
  __shared__ __align__(16) _Float16 Bsh[64][32];
  const int t    = threadIdx.x;
  const int wave = t >> 5;
  const int lane = t & 31;
  const int m16  = lane & 15;
  const int hi   = lane >> 4;
  const int K    = 38400;

  const int nblock = blockIdx.x * 64;           // 8 blocks over N
  const int kbase  = blockIdx.y * 48 * 32;      // 25 K-chunks of 48 steps

  v8f acc[4] = {};
  const int aoff = (wave * 16 + m16) * K + hi * 8;  // per-thread, loop-invariant

  // per-thread B-staging roles: LDS slot + 32-bit global element offset
  int bnl[4], bkq[4], boff[4];
  #pragma unroll
  for (int j = 0; j < 4; ++j) {
    int gi = j * 128 + t;            // 0..511
    bnl[j] = gi >> 3;
    bkq[j] = (gi & 7) * 4;
    boff[j] = (nblock + bnl[j]) * K + bkq[j];
  }

  float4 pre[4];
  {
    const float* Wk = W + kbase;
    #pragma unroll
    for (int j = 0; j < 4; ++j)
      pre[j] = *(const float4*)(Wk + boff[j]);
  }

  for (int ks = 0; ks < 48; ++ks) {
    const int k0 = kbase + ks * 32;

    // commit prefetched B tile to LDS
    #pragma unroll
    for (int j = 0; j < 4; ++j) {
      uint2 pkd;
      pkd.x = pk2(pre[j].x, pre[j].y);
      pkd.y = pk2(pre[j].z, pre[j].w);
      *(uint2*)&Bsh[bnl[j]][bkq[j]] = pkd;
    }
    __syncthreads();

    // prefetch next step's B tile (overlaps with fragment loads + WMMA)
    if (ks + 1 < 48) {
      const float* Wk = W + (k0 + 32);
      #pragma unroll
      for (int j = 0; j < 4; ++j)
        pre[j] = *(const float4*)(Wk + boff[j]);
    }

    // A fragment: two b128 loads from uniform base (sk + k0) + fixed offset
    Frag16 a;
    {
      const _Float16* skk = sk + k0;
      a.q[0] = *(const uint4*)(skk + aoff);
      a.q[1] = *(const uint4*)(skk + aoff + 16);
    }

    #pragma unroll
    for (int tn = 0; tn < 4; ++tn) {
      Frag16 bf;
      int nl = tn * 16 + m16;
      bf.q[0] = *(const uint4*)&Bsh[nl][hi * 8];
      bf.q[1] = *(const uint4*)&Bsh[nl][16 + hi * 8];
      acc[tn] = __builtin_amdgcn_wmma_f32_16x16x32_f16(
          false, a.h, false, bf.h, (short)0, acc[tn], false, false);
    }
    __syncthreads();
  }

  #pragma unroll
  for (int tn = 0; tn < 4; ++tn) {
    int n = nblock + tn * 16 + m16;
    #pragma unroll
    for (int r8 = 0; r8 < 8; ++r8) {
      int m = wave * 16 + r8 + hi * 8;
      atomicAdd(&h1[(size_t)m * 512 + n], acc[tn][r8]);
    }
  }
}

// ---------------------------------------------------------------------------
// Kernel 5: bias + LayerNorm(512) + fc2 (6x512) -> score (64,6)
// ---------------------------------------------------------------------------
__global__ __launch_bounds__(256)
void head_kernel(const float* __restrict__ h1, const float* __restrict__ fb1,
                 const float* __restrict__ n2w, const float* __restrict__ n2b,
                 const float* __restrict__ fw2, const float* __restrict__ fb2,
                 float* __restrict__ out) {
  __shared__ float vv[512];
  __shared__ float red[256];
  __shared__ float s_mu, s_rstd;
  const int r = blockIdx.x, t = threadIdx.x;

  float a0 = h1[r * 512 + t]       + fb1[t];
  float a1 = h1[r * 512 + 256 + t] + fb1[256 + t];

  red[t] = a0 + a1; __syncthreads();
  for (int s = 128; s > 0; s >>= 1) { if (t < s) red[t] += red[t + s]; __syncthreads(); }
  if (t == 0) s_mu = red[0] / 512.f;
  __syncthreads();
  red[t] = a0 * a0 + a1 * a1; __syncthreads();
  for (int s = 128; s > 0; s >>= 1) { if (t < s) red[t] += red[t + s]; __syncthreads(); }
  if (t == 0) s_rstd = rsqrtf(red[0] / 512.f - s_mu * s_mu + 1e-5f);
  __syncthreads();

  float mu = s_mu, rstd = s_rstd;
  vv[t]       = (a0 - mu) * rstd * n2w[t]       + n2b[t];
  vv[t + 256] = (a1 - mu) * rstd * n2w[t + 256] + n2b[t + 256];
  __syncthreads();

  if (t < 6) {
    float s = fb2[t];
    for (int c = 0; c < 512; ++c) s += vv[c] * fw2[t * 512 + c];
    out[r * 6 + t] = s;
  }
}

// ---------------------------------------------------------------------------
extern "C" void kernel_launch(void* const* d_in, const int* in_sizes, int n_in,
                              void* d_out, int out_size, void* d_ws, size_t ws_size,
                              hipStream_t stream) {
  const float*     x4   = (const float*)d_in[0];
  const float*     x5   = (const float*)d_in[1];
  const float*     boxes= (const float*)d_in[2];
  const float*     skel = (const float*)d_in[3];
  const int*       group= (const int*)d_in[4];
  /* d_in[5] real_in_num: unused by reference */
  const long long* labs = (const long long*)d_in[6];
  const float*     c1w  = (const float*)d_in[7];
  const float*     c1b  = (const float*)d_in[8];
  const float*     w2   = (const float*)d_in[9];
  const float*     b2   = (const float*)d_in[10];
  const float*     w3   = (const float*)d_in[11];
  const float*     b3   = (const float*)d_in[12];
  const float*     n1w  = (const float*)d_in[13];
  const float*     n1b  = (const float*)d_in[14];
  const float*     n2w  = (const float*)d_in[15];
  const float*     n2b  = (const float*)d_in[16];
  const float*     fw1  = (const float*)d_in[17];
  const float*     fb1  = (const float*)d_in[18];
  const float*     fw2  = (const float*)d_in[19];
  const float*     fb2  = (const float*)d_in[20];

  // workspace carve-up (all offsets multiples of 16 bytes)
  float*    f4c = (float*)d_ws;                 //  8*512*196   = 802816 f
  float*    f5c = f4c + 802816;                 //  8*1024*49   = 401408 f
  float*    app = f5c + 401408;                 //  8*1536*784  = 9633792 f
  float*    h1  = app + 9633792;                //  64*512      = 32768 f
  _Float16* skb = (_Float16*)(h1 + 32768);      //  64*38400 halves

  // conv2: C=1024 P=196 N=512 ; conv3: C=2048 P=49 N=1024
  conv1x1_wmma<<<dim3(13, 2, 8), 128, 0, stream>>>(x4, w2, b2, f4c, 1024, 196, 512);
  conv1x1_wmma<<<dim3(4, 4, 8),  128, 0, stream>>>(x5, w3, b3, f5c, 2048, 49, 1024);

  int total = 8 * 1536 * 28 * 28;
  resize_concat<<<(total + 255) / 256, 256, 0, stream>>>(f4c, f5c, app, total);

  roi_ln1<<<64, 256, 38400 * sizeof(float), stream>>>(
      app, boxes, skel, group, labs, c1w, c1b, n1w, n1b, skb);

  (void)hipMemsetAsync(h1, 0, 64 * 512 * sizeof(float), stream);
  fc1_wmma<<<dim3(8, 25), 128, 0, stream>>>(skb, fw1, h1);

  head_kernel<<<64, 256, 0, stream>>>(h1, fb1, n2w, n2b, fw2, fb2, (float*)d_out);
}
